// EEG_LSTM_GAT_Model_6708738916893
// MI455X (gfx1250) — compile-verified, matrix operand order checked by reading, not run on hardware
//
#include <hip/hip_runtime.h>
#include <hip/hip_bf16.h>

// ---------------------------------------------------------------------------
// EEG LSTM + GATv2 forward for MI455X (gfx1250), wave32, WMMA-based.
//
//   B=64 batch, T=2048 time, E=19 electrodes, H=32 hidden/dir, 4H=128 gates
//   HEADS=8, C=64 -> 512 GAT features
//
// The model is recurrence-latency bound (T=2048 serial steps, ~1 MFLOP/step);
// design goal: shortest per-step critical path.
//   - W_hh B-fragments VGPR-resident for the whole T loop
//   - h (f16) and gate pre-activations round-trip through LDS only
//   - gate bias folded into the WMMA C-operand (broadcast fragment)
//   - sigmoid/tanh via __expf + v_rcp_f32 (TRANS ops, no IEEE divide)
//   - next x column prefetched (10 MB stream, L2-resident in 192 MB L2)
// ---------------------------------------------------------------------------

#define NE    19
#define NH    32
#define NB    64
#define NT    2048
#define NG    128      // 4H
#define ROWS  1216     // B*E
#define FDIM  512      // HEADS*C

typedef __attribute__((ext_vector_type(16))) _Float16 v16h;
typedef __attribute__((ext_vector_type(8)))  float    v8f;

union V16H {
    v16h   v;
    _Float16 h[16];
    float4 f4[2];
};

__device__ __forceinline__ float fast_sigmoid(float x) {
    return __builtin_amdgcn_rcpf(1.0f + __expf(-x));
}
__device__ __forceinline__ float fast_tanh(float x) {
    // tanh(x) = 2*sigmoid(2x) - 1
    return fmaf(2.0f, __builtin_amdgcn_rcpf(1.0f + __expf(-2.0f * x)), -1.0f);
}

// ---------------------------------------------------------------------------
// 1) Per-(b,e) normalization over time; writes xn transposed to [E][T][B]
// ---------------------------------------------------------------------------
__global__ __launch_bounds__(256) void norm_kernel(
    const float* __restrict__ x,   // [B, T, E]
    float* __restrict__ xn)        // [E, T, B]
{
    const int b = blockIdx.x / NE;
    const int e = blockIdx.x % NE;
    __shared__ float red[256];

    float s = 0.f, s2 = 0.f;
    for (int t = threadIdx.x; t < NT; t += 256) {
        float v = x[((size_t)b * NT + t) * NE + e];
        s += v; s2 += v * v;
    }
    red[threadIdx.x] = s; __syncthreads();
    for (int o = 128; o > 0; o >>= 1) {
        if (threadIdx.x < o) red[threadIdx.x] += red[threadIdx.x + o];
        __syncthreads();
    }
    const float mean = red[0] / (float)NT;
    __syncthreads();
    red[threadIdx.x] = s2; __syncthreads();
    for (int o = 128; o > 0; o >>= 1) {
        if (threadIdx.x < o) red[threadIdx.x] += red[threadIdx.x + o];
        __syncthreads();
    }
    const float var = (red[0] - (float)NT * mean * mean) / (float)(NT - 1); // unbiased
    const float inv = 1.0f / (sqrtf(fmaxf(var, 0.f)) + 1e-6f);

    for (int t = threadIdx.x; t < NT; t += 256) {
        float v = x[((size_t)b * NT + t) * NE + e];
        xn[((size_t)e * NT + t) * NB + b] = (v - mean) * inv;
    }
}

// ---------------------------------------------------------------------------
// 2) Per-(electrode, direction) LSTM. Block = 256 threads = 8 waves.
//    Wave w owns gate columns [16w, 16w+16); per step 4 WMMAs (16x16x32 f16)
//    cover all 64 batch rows; the C-operand is a broadcast bias fragment.
//
//    A-fragment layout (ISA 7.12.2, 16-bit A 16x32):
//      lanes 0-15 : halves 0..7 = K0..7,  halves 8..15 = K16..23
//      lanes 16-31: halves 0..7 = K8..15, halves 8..15 = K24..31
//    B-fragment mirrored (lane%16 = column, same K interleave).
// ---------------------------------------------------------------------------
__global__ __launch_bounds__(256) void lstm_kernel(
    const float* __restrict__ xn,       // [E, T, B]
    const float* __restrict__ Wih_f,    // [E, 128]
    const float* __restrict__ Whh_f,    // [E, 128, 32]
    const float* __restrict__ bih_f,    // [E, 128]
    const float* __restrict__ bhh_f,    // [E, 128]
    const float* __restrict__ Wih_b,
    const float* __restrict__ Whh_b,
    const float* __restrict__ bih_b,
    const float* __restrict__ bhh_b,
    float* __restrict__ feat)           // [B, E, 64]  (dir0 -> [0:32), dir1 -> [32:64))
{
    __shared__ _Float16 sH[NB * NH];    // 4 KB   h (f16) for A-fragments
    __shared__ float    sG[NB * NG];    // 32 KB  gate pre-activations (incl. bias)
    __shared__ float    sWih[NG];

    const int e   = blockIdx.x;
    const int dir = blockIdx.y;
    const float* Wih = dir ? Wih_b : Wih_f;
    const float* Whh = dir ? Whh_b : Whh_f;
    const float* bih = dir ? bih_b : bih_f;
    const float* bhh = dir ? bhh_b : bhh_f;

    const int tid  = threadIdx.x;
    const int wave = tid >> 5;
    const int lane = tid & 31;
    const int nl   = lane & 15;
    const int hi   = lane >> 4;
    const int koff = hi ? 8 : 0;

    if (tid < NG) sWih[tid] = Wih[e * NG + tid];
    for (int idx = tid; idx < NB * NH; idx += 256) sH[idx] = (_Float16)0.f;

    // Resident B-fragment: column = 16*wave + nl, B[k][n] = W_hh[e][n][k]
    V16H bfrag;
    v8f  cinit;                          // bias broadcast into all 8 C rows
    {
        const int ncol = 16 * wave + nl;
        const float* wrow = Whh + ((size_t)e * NG + ncol) * NH;
        #pragma unroll
        for (int v = 0; v < 8; ++v) {
            int k0 = ((v < 4) ? 2 * v : 2 * v + 8) + koff;
            bfrag.h[2 * v]     = (_Float16)wrow[k0];
            bfrag.h[2 * v + 1] = (_Float16)wrow[k0 + 1];
        }
        const float bv = bih[e * NG + ncol] + bhh[e * NG + ncol];
        #pragma unroll
        for (int r = 0; r < 8; ++r) cinit[r] = bv;
    }

    // Thread-owned cell state: b = tid/4, hidden = (tid%4)*8 + q
    const int bown  = tid >> 2;
    const int hbase = (tid & 3) * 8;
    float creg[8], hreg[8];
    #pragma unroll
    for (int q = 0; q < 8; ++q) { creg[q] = 0.f; hreg[q] = 0.f; }

    // direction as pointer + signed stride (no per-step selects)
    const float* xptr = xn + (size_t)e * NT * NB + (dir ? (size_t)(NT - 1) * NB : 0);
    const ptrdiff_t xstep = dir ? -(ptrdiff_t)NB : (ptrdiff_t)NB;

    __syncthreads();

    for (int step = 0; step < NT; ++step) {
        // --- WMMA phase: G = H @ W_hh^T + bias, EXEC all ones ---
        #pragma unroll
        for (int m = 0; m < 4; ++m) {
            V16H afrag;
            const _Float16* hrow = sH + (m * 16 + nl) * NH + koff;
            afrag.f4[0] = *(const float4*)(hrow);        // K koff..koff+7
            afrag.f4[1] = *(const float4*)(hrow + 16);   // K koff+16..koff+23
            v8f acc = __builtin_amdgcn_wmma_f32_16x16x32_f16(
                false, afrag.v, false, bfrag.v, (short)0, cinit, false, false);
            #pragma unroll
            for (int r = 0; r < 8; ++r)
                sG[(m * 16 + r + 8 * hi) * NG + 16 * wave + nl] = acc[r];
        }
        __syncthreads();

        // --- gate phase: each thread updates 8 (b, hidden) cells ---
        const float xval = xptr[bown];
        __builtin_prefetch(xptr + xstep + bown, 0, 1);   // speculative, safe
        xptr += xstep;
        const float* grow = sG + bown * NG;
        #pragma unroll
        for (int q = 0; q < 8; ++q) {
            const int hx = hbase + q;
            float gi = fmaf(xval, sWih[hx],      grow[hx]);
            float gf = fmaf(xval, sWih[32 + hx], grow[32 + hx]);
            float gg = fmaf(xval, sWih[64 + hx], grow[64 + hx]);
            float go = fmaf(xval, sWih[96 + hx], grow[96 + hx]);
            float c  = fast_sigmoid(gf) * creg[q] + fast_sigmoid(gi) * fast_tanh(gg);
            creg[q]  = c;
            float h  = fast_sigmoid(go) * fast_tanh(c);
            hreg[q]  = h;
            sH[bown * NH + hx] = (_Float16)h;
        }
        __syncthreads();
    }

    // final hidden -> feat[b][e][dir*32 + hidden]
    #pragma unroll
    for (int q = 0; q < 8; ++q)
        feat[((size_t)bown * NE + e) * 64 + dir * NH + hbase + q] = hreg[q];
}

// ---------------------------------------------------------------------------
// 3) Per-electrode LayerNorm over 2H=64, emit f16 for the WMMA GEMM.
// ---------------------------------------------------------------------------
__global__ __launch_bounds__(64) void ln_kernel(
    const float* __restrict__ feat,   // [ROWS, 64] (row = b*E + e)
    const float* __restrict__ gamma,  // [E, 64]
    const float* __restrict__ beta,   // [E, 64]
    _Float16* __restrict__ nf16)      // [ROWS, 64]
{
    const int be = blockIdx.x;
    const int e  = be % NE;
    const int c  = threadIdx.x;
    __shared__ float red[64];

    const float v = feat[(size_t)be * 64 + c];
    red[c] = v; __syncthreads();
    for (int o = 32; o > 0; o >>= 1) {
        if (c < o) red[c] += red[c + o];
        __syncthreads();
    }
    const float m = red[0] / 64.f;
    __syncthreads();
    const float d = v - m;
    red[c] = d * d; __syncthreads();
    for (int o = 32; o > 0; o >>= 1) {
        if (c < o) red[c] += red[c + o];
        __syncthreads();
    }
    const float var = red[0] / 64.f;
    const float nf  = d * rsqrtf(var + 1e-5f) * gamma[e * 64 + c] + beta[e * 64 + c];
    nf16[(size_t)be * 64 + c] = (_Float16)nf;
}

// ---------------------------------------------------------------------------
// 4) WMMA GEMM: xl/xr[1216,512] = nf16[1216,64] @ W^T[64,512] + b
//    Grid (76, 4, 2): M-tile, N-group of 8 waves, {W_l, W_r} select.
//    K=64 -> two chained 16x16x32 WMMAs; bias rides in the first C-operand.
// ---------------------------------------------------------------------------
__global__ __launch_bounds__(256) void gat_lin_kernel(
    const _Float16* __restrict__ nf16,
    const float* __restrict__ W_l, const float* __restrict__ b_l,
    const float* __restrict__ W_r, const float* __restrict__ b_r,
    float* __restrict__ xl, float* __restrict__ xr)
{
    const int mt  = blockIdx.x;          // 0..75  (1216 = 76*16 exactly)
    const int ngp = blockIdx.y;          // 0..3
    const int sel = blockIdx.z;          // 0 -> l, 1 -> r
    const float* W    = sel ? W_r : W_l;
    const float* bias = sel ? b_r : b_l;
    float*       out  = sel ? xr  : xl;

    const int wave = threadIdx.x >> 5;
    const int lane = threadIdx.x & 31;
    const int nl   = lane & 15;
    const int hi   = lane >> 4;
    const int koff = hi ? 8 : 0;
    const int nt   = ngp * 8 + wave;     // 0..31
    const int ncol = nt * 16 + nl;       // 0..511

    V16H b0, b1;
    const float* wrow = W + (size_t)ncol * 64;   // W[n][k]
    #pragma unroll
    for (int v = 0; v < 8; ++v) {
        int k0 = ((v < 4) ? 2 * v : 2 * v + 8) + koff;
        b0.h[2 * v]     = (_Float16)wrow[k0];
        b0.h[2 * v + 1] = (_Float16)wrow[k0 + 1];
        b1.h[2 * v]     = (_Float16)wrow[32 + k0];
        b1.h[2 * v + 1] = (_Float16)wrow[32 + k0 + 1];
    }

    const _Float16* arow = nf16 + (size_t)(mt * 16 + nl) * 64;
    V16H a0, a1;
    a0.f4[0] = *(const float4*)(arow + koff);
    a0.f4[1] = *(const float4*)(arow + koff + 16);
    a1.f4[0] = *(const float4*)(arow + 32 + koff);
    a1.f4[1] = *(const float4*)(arow + 32 + koff + 16);

    v8f acc;
    const float bv = bias[ncol];
    #pragma unroll
    for (int r = 0; r < 8; ++r) acc[r] = bv;
    acc = __builtin_amdgcn_wmma_f32_16x16x32_f16(false, a0.v, false, b0.v, (short)0, acc, false, false);
    acc = __builtin_amdgcn_wmma_f32_16x16x32_f16(false, a1.v, false, b1.v, (short)0, acc, false, false);

    #pragma unroll
    for (int r = 0; r < 8; ++r)
        out[(size_t)(mt * 16 + r + 8 * hi) * FDIM + ncol] = acc[r];
}

// ---------------------------------------------------------------------------
// 5) Dense GATv2 attention, one block per (b, dst i, head). 64 threads.
// ---------------------------------------------------------------------------
__global__ __launch_bounds__(64) void gat_att_kernel(
    const float* __restrict__ xl,       // [ROWS, 512]
    const float* __restrict__ xr,       // [ROWS, 512]
    const float* __restrict__ att,      // [HEADS, 64]
    const float* __restrict__ gat_bias, // [512]
    float* __restrict__ gout)           // [ROWS, 512]
{
    const int hd = blockIdx.x & 7;
    const int bi = blockIdx.x >> 3;     // b*E + i
    const int b  = bi / NE;
    const int c  = threadIdx.x;

    __shared__ float xls[NE][64];
    __shared__ float xrs[64];
    __shared__ float atts[64];
    __shared__ float es[NE];

    xrs[c]  = xr[(size_t)bi * FDIM + hd * 64 + c];
    atts[c] = att[hd * 64 + c];
    for (int j = 0; j < NE; ++j)
        xls[j][c] = xl[(size_t)(b * NE + j) * FDIM + hd * 64 + c];
    __syncthreads();

    if (c < NE) {
        float s = 0.f;
        for (int k = 0; k < 64; ++k) {
            float p = xls[c][k] + xrs[k];
            p = (p > 0.f) ? p : 0.2f * p;      // leaky_relu(0.2)
            s += p * atts[k];
        }
        es[c] = s;
    }
    __syncthreads();

    float mx = -1e30f;
    for (int j = 0; j < NE; ++j) mx = fmaxf(mx, es[j]);
    float den = 0.f;
    for (int j = 0; j < NE; ++j) den += __expf(es[j] - mx);
    float o = 0.f;
    for (int j = 0; j < NE; ++j) o += __expf(es[j] - mx) * xls[j][c];
    gout[(size_t)bi * FDIM + hd * 64 + c] = o * __builtin_amdgcn_rcpf(den) + gat_bias[hd * 64 + c];
}

// ---------------------------------------------------------------------------
// 6) Mean-pool over electrodes + FC -> logits [B]
// ---------------------------------------------------------------------------
__global__ __launch_bounds__(256) void pool_fc_kernel(
    const float* __restrict__ gout,   // [ROWS, 512]
    const float* __restrict__ W_fc,   // [1, 512]
    const float* __restrict__ b_fc,   // [1]
    float* __restrict__ out)          // [B]
{
    const int b = blockIdx.x;
    __shared__ float red[256];
    float acc = 0.f;
    for (int f = threadIdx.x; f < FDIM; f += 256) {
        float s = 0.f;
        for (int i = 0; i < NE; ++i)
            s += gout[(size_t)(b * NE + i) * FDIM + f];
        acc += (s / (float)NE) * W_fc[f];
    }
    red[threadIdx.x] = acc; __syncthreads();
    for (int o = 128; o > 0; o >>= 1) {
        if (threadIdx.x < o) red[threadIdx.x] += red[threadIdx.x + o];
        __syncthreads();
    }
    if (threadIdx.x == 0) out[b] = red[0] + b_fc[0];
}

// ---------------------------------------------------------------------------
extern "C" void kernel_launch(void* const* d_in, const int* in_sizes, int n_in,
                              void* d_out, int out_size, void* d_ws, size_t ws_size,
                              hipStream_t stream) {
    (void)in_sizes; (void)n_in; (void)out_size; (void)ws_size;

    const float* x      = (const float*)d_in[0];
    const float* Wih_f  = (const float*)d_in[1];
    const float* Whh_f  = (const float*)d_in[2];
    const float* bih_f  = (const float*)d_in[3];
    const float* bhh_f  = (const float*)d_in[4];
    const float* Wih_b  = (const float*)d_in[5];
    const float* Whh_b  = (const float*)d_in[6];
    const float* bih_b  = (const float*)d_in[7];
    const float* bhh_b  = (const float*)d_in[8];
    const float* ln_g   = (const float*)d_in[9];
    const float* ln_b   = (const float*)d_in[10];
    const float* W_l    = (const float*)d_in[11];
    const float* b_l    = (const float*)d_in[12];
    const float* W_r    = (const float*)d_in[13];
    const float* b_r    = (const float*)d_in[14];
    const float* att    = (const float*)d_in[15];
    const float* g_bias = (const float*)d_in[16];
    const float* W_fc   = (const float*)d_in[17];
    const float* b_fc   = (const float*)d_in[18];
    float* out = (float*)d_out;

    // workspace carve-up (all offsets 256B-aligned)
    char* ws = (char*)d_ws;
    size_t off = 0;
    float* xn = (float*)(ws + off);         off += (size_t)NE * NT * NB * sizeof(float);  // 9.96 MB
    float* feat = (float*)(ws + off);       off += (size_t)ROWS * 64 * sizeof(float);     // 311 KB
    _Float16* nf16 = (_Float16*)(ws + off); off += (size_t)ROWS * 64 * sizeof(_Float16);  // 156 KB
    float* xl = (float*)(ws + off);         off += (size_t)ROWS * FDIM * sizeof(float);   // 2.49 MB
    float* xr = (float*)(ws + off);         off += (size_t)ROWS * FDIM * sizeof(float);   // 2.49 MB
    float* gout = (float*)(ws + off);       off += (size_t)ROWS * FDIM * sizeof(float);   // 2.49 MB

    norm_kernel<<<ROWS, 256, 0, stream>>>(x, xn);

    lstm_kernel<<<dim3(NE, 2), 256, 0, stream>>>(
        xn, Wih_f, Whh_f, bih_f, bhh_f, Wih_b, Whh_b, bih_b, bhh_b, feat);

    ln_kernel<<<ROWS, 64, 0, stream>>>(feat, ln_g, ln_b, nf16);

    gat_lin_kernel<<<dim3(76, 4, 2), 256, 0, stream>>>(
        nf16, W_l, b_l, W_r, b_r, xl, xr);

    gat_att_kernel<<<ROWS * 8, 64, 0, stream>>>(xl, xr, att, g_bias, gout);

    pool_fc_kernel<<<NB, 256, 0, stream>>>(gout, W_fc, b_fc, out);
}
